// GVPAdapter_74148315398388
// MI455X (gfx1250) — matrix operand order repaired and checked by reading, not compiled, not run on hardware
//
#include <hip/hip_runtime.h>
#include <hip/hip_bf16.h>
#include <cstdint>
#include <cstddef>

// Problem constants (match reference)
#define B_   8
#define NL_  1024
#define NP_  2048
#define D_   512
#define D1_  513          // D+1 (scalar channel count into GVP s_proj/gate)
#define DP_  544          // D+1 padded up to a multiple of 32 for the WMMA K loop

// Wave-level register blocking: each wave owns a (WM*16) x (WN*16) output tile.
// 2x4 => 8 WMMAs per 6 fragment loads per k-step (~2.7x intensity of 1x1).
#define WM 2
#define WN 4

typedef float  v8f   __attribute__((ext_vector_type(8)));
typedef __bf16 v16bf __attribute__((ext_vector_type(16)));
typedef unsigned int u32x4 __attribute__((ext_vector_type(4)));

union FragU { u32x4 q[2]; v16bf v; };

__device__ __forceinline__ unsigned short f2bf(float f) {
  unsigned u = __builtin_bit_cast(unsigned, f);
  u += 0x7FFFu + ((u >> 16) & 1u);           // round-to-nearest-even
  return (unsigned short)(u >> 16);
}
__device__ __forceinline__ float bf2f(unsigned short h) {
  return __builtin_bit_cast(float, (unsigned)h << 16);
}

// ---------------------------------------------------------------------------
// Register-blocked WMMA tile: acc[WM][WN] (each 16x16 f32) accumulates
// A(32xK,bf16) * B(Kx64,bf16).  A is MxK row-major; Bm is NxK row-major
// (weight / "B^T" layout).  Fragment layouts per CDNA5 ISA 7.12.2:
//   A: lane l<16 row M=l holds K {0..7,16..23}; lane 16+l holds {8..15,24..31}
//   B: lane n<16 col N=n holds K {0..7,16..23}; lane 16+n holds {8..15,24..31}
// ---------------------------------------------------------------------------
__device__ __forceinline__ void gemm_tile_blk(
    const unsigned short* __restrict__ A,
    const unsigned short* __restrict__ Bm,
    int lda, int ldb, int K, int tileM, int tileN, int lane,
    v8f acc[WM][WN])
{
  const int l15 = lane & 15;
  const int hi  = lane >> 4;
  const unsigned short* ap[WM];
  const unsigned short* bp[WN];
#pragma unroll
  for (int mi = 0; mi < WM; ++mi)
    ap[mi] = A + (size_t)(tileM + mi * 16 + l15) * lda + (hi << 3);
#pragma unroll
  for (int ni = 0; ni < WN; ++ni)
    bp[ni] = Bm + (size_t)(tileN + ni * 16 + l15) * ldb + (hi << 3);

  const v8f vzero = {0.f, 0.f, 0.f, 0.f, 0.f, 0.f, 0.f, 0.f};
#pragma unroll
  for (int mi = 0; mi < WM; ++mi)
#pragma unroll
    for (int ni = 0; ni < WN; ++ni)
      acc[mi][ni] = vzero;

  for (int k = 0; k < K; k += 32) {
    FragU a[WM], b[WN];
#pragma unroll
    for (int mi = 0; mi < WM; ++mi) {
      a[mi].q[0] = *(const u32x4*)(ap[mi] + k);
      a[mi].q[1] = *(const u32x4*)(ap[mi] + k + 16);
      __builtin_prefetch(ap[mi] + k + 128, 0, 3);   // near-cache lookahead
    }
#pragma unroll
    for (int ni = 0; ni < WN; ++ni) {
      b[ni].q[0] = *(const u32x4*)(bp[ni] + k);
      b[ni].q[1] = *(const u32x4*)(bp[ni] + k + 16);
      __builtin_prefetch(bp[ni] + k + 128, 0, 3);
    }
#pragma unroll
    for (int mi = 0; mi < WM; ++mi)
#pragma unroll
      for (int ni = 0; ni < WN; ++ni)
        acc[mi][ni] = __builtin_amdgcn_wmma_f32_16x16x32_bf16(
            false, a[mi].v, false, b[ni].v, (short)0, acc[mi][ni],
            false, false);
  }
}

// Block = 256 threads = 8 waves stacked along M; block covers 256(M) x 64(N).
#define TILE_COORDS()                                        \
  const int lane  = threadIdx.x & 31;                        \
  const int wave  = threadIdx.x >> 5;                        \
  const int tileM = (blockIdx.y * 8 + wave) * (WM * 16);     \
  const int tileN = blockIdx.x * (WN * 16);                  \
  const int l15   = lane & 15;                               \
  const int hi    = lane >> 4;

// Iterate all (row,col,value) of the wave tile in the epilogue.
// Variadic so commas inside the body don't split macro arguments.
#define EPILOGUE_LOOP(...)                                   \
  _Pragma("unroll")                                          \
  for (int mi = 0; mi < WM; ++mi)                            \
    _Pragma("unroll")                                        \
    for (int ni = 0; ni < WN; ++ni) {                        \
      const int col = tileN + ni * 16 + l15;                 \
      _Pragma("unroll")                                      \
      for (int j = 0; j < 8; ++j) {                          \
        const int row = tileM + mi * 16 + hi * 8 + j;        \
        const float cv = acc[mi][ni][j];                     \
        { __VA_ARGS__ }                                      \
      }                                                      \
    }

// ---------------------------------------------------------------------------
// Converts
// ---------------------------------------------------------------------------
__global__ void k_f32_to_bf16(const float* __restrict__ s,
                              unsigned short* __restrict__ d, int n) {
  for (int i = blockIdx.x * blockDim.x + threadIdx.x; i < n;
       i += gridDim.x * blockDim.x)
    d[i] = f2bf(s[i]);
}

// s1_w: (D, D+1) -> bf16 (D, DP_) zero-padded
__global__ void k_pad_s1(const float* __restrict__ w,
                         unsigned short* __restrict__ d) {
  const int total = D_ * DP_;
  for (int i = blockIdx.x * blockDim.x + threadIdx.x; i < total;
       i += gridDim.x * blockDim.x) {
    int n = i / DP_, k = i - n * DP_;
    d[i] = (k < D1_) ? f2bf(w[n * D1_ + k]) : (unsigned short)0;
  }
}

// ---------------------------------------------------------------------------
// GEMM variants (epilogue fused)
// ---------------------------------------------------------------------------

// C = A * W^T + bias, bf16 out (Q and K projections)
__global__ void k_gemm_bf16out(const unsigned short* __restrict__ A,
                               const unsigned short* __restrict__ Bm,
                               const float* __restrict__ bias,
                               unsigned short* __restrict__ Cbf,
                               int K, int lda, int ldb, int ldc) {
  TILE_COORDS();
  v8f acc[WM][WN];
  gemm_tile_blk(A, Bm, lda, ldb, K, tileM, tileN, lane, acc);
  EPILOGUE_LOOP(
    Cbf[(size_t)row * ldc + col] = f2bf(cv + bias[col]);
  )
}

// V projection, stored TRANSPOSED per batch: Vt[b][d][p]  (so probs@V reuses
// the same N-major fragment loader)
__global__ void k_gemm_vT(const unsigned short* __restrict__ A,
                          const unsigned short* __restrict__ Bm,
                          const float* __restrict__ bias,
                          unsigned short* __restrict__ Vt) {
  TILE_COORDS();
  v8f acc[WM][WN];
  gemm_tile_blk(A, Bm, D_, D_, D_, tileM, tileN, lane, acc);
  EPILOGUE_LOOP(
    int b = row >> 11;                     // row = flat b*NP + p
    int p = row & (NP_ - 1);
    Vt[((size_t)b * D_ + col) * NP_ + p] = f2bf(cv + bias[col]);
  )
}

// scores[b] = Q[b] @ K[b]^T / sqrt(D) + mask_bias(dist), f32 out
__global__ void k_gemm_scores(const unsigned short* __restrict__ Qbf,
                              const unsigned short* __restrict__ Kbf,
                              const float* __restrict__ posL,
                              const float* __restrict__ posP,
                              float* __restrict__ scores) {
  TILE_COORDS();
  const int b = blockIdx.z;
  const unsigned short* A  = Qbf + (size_t)b * NL_ * D_;
  const unsigned short* Bm = Kbf + (size_t)b * NP_ * D_;
  float* C = scores + (size_t)b * NL_ * NP_;
  v8f acc[WM][WN];
  gemm_tile_blk(A, Bm, D_, D_, D_, tileM, tileN, lane, acc);
  const float invScale = 0.04419417382415922f;   // 1/sqrt(512)
  EPILOGUE_LOOP(
    float dx = posL[((size_t)b * NL_ + row) * 3 + 0]
             - posP[((size_t)b * NP_ + col) * 3 + 0];
    float dy = posL[((size_t)b * NL_ + row) * 3 + 1]
             - posP[((size_t)b * NP_ + col) * 3 + 1];
    float dz = posL[((size_t)b * NL_ + row) * 3 + 2]
             - posP[((size_t)b * NP_ + col) * 3 + 2];
    float dist = sqrtf(dx * dx + dy * dy + dz * dz);
    C[(size_t)row * NP_ + col] = cv * invScale + (dist > 10.f ? -1e9f : 0.f);
  )
}

// s_context[b] = probs[b] @ V[b]; fused residual x_L; writes s_comb cols 0..511
__global__ void k_gemm_ctx(const unsigned short* __restrict__ probsBf,
                           const unsigned short* __restrict__ Vt,
                           const float* __restrict__ xL,
                           unsigned short* __restrict__ scomb) {
  TILE_COORDS();
  const int b = blockIdx.z;
  const unsigned short* A  = probsBf + (size_t)b * NL_ * NP_;
  const unsigned short* Bm = Vt      + (size_t)b * D_ * NP_;
  v8f acc[WM][WN];
  gemm_tile_blk(A, Bm, NP_, NP_, NP_, tileM, tileN, lane, acc);
  EPILOGUE_LOOP(
    size_t g = (size_t)b * NL_ + row;
    scomb[g * DP_ + col] = f2bf(cv + xL[g * D_ + col]);
  )
}

// h = silu(s_comb @ s1_w^T + s1_b), bf16 out
__global__ void k_gemm_silu(const unsigned short* __restrict__ A,
                            const unsigned short* __restrict__ Bm,
                            const float* __restrict__ bias,
                            unsigned short* __restrict__ H) {
  TILE_COORDS();
  v8f acc[WM][WN];
  gemm_tile_blk(A, Bm, DP_, DP_, DP_, tileM, tileN, lane, acc);
  EPILOGUE_LOOP(
    float t = cv + bias[col];
    H[(size_t)row * D_ + col] = f2bf(t / (1.f + expf(-t)));   // silu
  )
}

// s_out = h @ s2_w^T + s2_b, f32 out -> d_out
__global__ void k_gemm_f32out(const unsigned short* __restrict__ A,
                              const unsigned short* __restrict__ Bm,
                              const float* __restrict__ bias,
                              float* __restrict__ C) {
  TILE_COORDS();
  v8f acc[WM][WN];
  gemm_tile_blk(A, Bm, D_, D_, D_, tileM, tileN, lane, acc);
  EPILOGUE_LOOP(
    C[(size_t)row * D_ + col] = cv + bias[col];
  )
}

// ---------------------------------------------------------------------------
// Softmax over NP=2048 per (b,nl) row; 256 threads, 8 elems/thread.
// Writes fp32 probs to d_out and bf16 probs for the probs@V WMMA.
// ---------------------------------------------------------------------------
__global__ void k_softmax(const float* __restrict__ scores,
                          float* __restrict__ probs,
                          unsigned short* __restrict__ probsBf) {
  const int row = blockIdx.x;
  const int t = threadIdx.x;
  const float* s = scores + (size_t)row * NP_;
  float v[8];
  float m = -3.4e38f;
  for (int i = 0; i < 8; ++i) { v[i] = s[t + i * 256]; m = fmaxf(m, v[i]); }
  __shared__ float red[256];
  red[t] = m; __syncthreads();
  for (int o = 128; o > 0; o >>= 1) {
    if (t < o) red[t] = fmaxf(red[t], red[t + o]);
    __syncthreads();
  }
  m = red[0]; __syncthreads();
  float sum = 0.f;
  for (int i = 0; i < 8; ++i) { v[i] = expf(v[i] - m); sum += v[i]; }
  red[t] = sum; __syncthreads();
  for (int o = 128; o > 0; o >>= 1) {
    if (t < o) red[t] += red[t + o];
    __syncthreads();
  }
  const float inv = 1.f / red[0];
  float* po = probs + (size_t)row * NP_;
  unsigned short* pb = probsBf + (size_t)row * NP_;
  for (int i = 0; i < 8; ++i) {
    float pv = v[i] * inv;
    po[t + i * 256] = pv;
    pb[t + i * 256] = f2bf(pv);
  }
}

// ---------------------------------------------------------------------------
// V_agg[b,nl,:] = sum_p probs * (pos_L-pos_P)/(dist+1e-6); also writes the
// v_norm column (512) of s_comb and zeroes pad columns 513..543.
// ---------------------------------------------------------------------------
__global__ void k_vagg(const float* __restrict__ probs,
                       const float* __restrict__ posL,
                       const float* __restrict__ posP,
                       const float* __restrict__ vprojw,
                       float* __restrict__ Vagg,
                       unsigned short* __restrict__ scomb) {
  const int row = blockIdx.x;               // b*NL + nl
  const int b = row / NL_;
  const int t = threadIdx.x;
  const float plx = posL[(size_t)row * 3 + 0];
  const float ply = posL[(size_t)row * 3 + 1];
  const float plz = posL[(size_t)row * 3 + 2];
  float sx = 0.f, sy = 0.f, sz = 0.f;
  for (int p = t; p < NP_; p += 256) {
    float pr = probs[(size_t)row * NP_ + p];
    float dx = plx - posP[((size_t)b * NP_ + p) * 3 + 0];
    float dy = ply - posP[((size_t)b * NP_ + p) * 3 + 1];
    float dz = plz - posP[((size_t)b * NP_ + p) * 3 + 2];
    float inv = 1.f / (sqrtf(dx * dx + dy * dy + dz * dz) + 1e-6f);
    sx += pr * dx * inv; sy += pr * dy * inv; sz += pr * dz * inv;
  }
  __shared__ float rx[256], ry[256], rz[256];
  rx[t] = sx; ry[t] = sy; rz[t] = sz; __syncthreads();
  for (int o = 128; o > 0; o >>= 1) {
    if (t < o) { rx[t] += rx[t + o]; ry[t] += ry[t + o]; rz[t] += rz[t + o]; }
    __syncthreads();
  }
  if (t == 0) {
    Vagg[row * 3 + 0] = rx[0];
    Vagg[row * 3 + 1] = ry[0];
    Vagg[row * 3 + 2] = rz[0];
    float vn = fabsf(vprojw[0]) *
               sqrtf(rx[0] * rx[0] + ry[0] * ry[0] + rz[0] * rz[0]);
    scomb[(size_t)row * DP_ + D_] = f2bf(vn);
  }
  if (t >= 1 && t < 32) scomb[(size_t)row * DP_ + D_ + t] = 0;  // pad cols
}

// ---------------------------------------------------------------------------
// gate = sigmoid(s_comb . gate_w + gate_b); V_out = V_agg * vproj_w * gate
// ---------------------------------------------------------------------------
__global__ void k_gate(const unsigned short* __restrict__ scomb,
                       const float* __restrict__ gatew,
                       const float* __restrict__ gateb,
                       const float* __restrict__ vprojw,
                       const float* __restrict__ Vagg,
                       float* __restrict__ vout) {
  const int row = blockIdx.x;
  const int t = threadIdx.x;                // 128 threads
  float dot = 0.f;
  for (int i = t; i < D1_; i += 128)
    dot += bf2f(scomb[(size_t)row * DP_ + i]) * gatew[i];
  __shared__ float red[128];
  red[t] = dot; __syncthreads();
  for (int o = 64; o > 0; o >>= 1) {
    if (t < o) red[t] += red[t + o];
    __syncthreads();
  }
  if (t < 3) {
    float g = 1.f / (1.f + expf(-(red[0] + gateb[0])));
    vout[row * 3 + t] = Vagg[row * 3 + t] * vprojw[0] * g;
  }
}

// ---------------------------------------------------------------------------
extern "C" void kernel_launch(void* const* d_in, const int* in_sizes, int n_in,
                              void* d_out, int out_size, void* d_ws, size_t ws_size,
                              hipStream_t stream) {
  (void)in_sizes; (void)n_in; (void)out_size; (void)ws_size;
  const float* xL     = (const float*)d_in[0];
  const float* xP     = (const float*)d_in[1];
  const float* posL   = (const float*)d_in[2];
  const float* posP   = (const float*)d_in[3];
  const float* Wq     = (const float*)d_in[4];
  const float* bq     = (const float*)d_in[5];
  const float* Wk     = (const float*)d_in[6];
  const float* bk     = (const float*)d_in[7];
  const float* Wv     = (const float*)d_in[8];
  const float* bv     = (const float*)d_in[9];
  const float* vprojw = (const float*)d_in[10];
  const float* s1w    = (const float*)d_in[11];
  const float* s1b    = (const float*)d_in[12];
  const float* s2w    = (const float*)d_in[13];
  const float* s2b    = (const float*)d_in[14];
  const float* gatew  = (const float*)d_in[15];
  const float* gateb  = (const float*)d_in[16];

  float* out  = (float*)d_out;
  float* outS = out;                                  // (B,NL,D)
  float* outV = outS + (size_t)B_ * NL_ * D_;         // (B,NL,1,3)
  float* outP = outV + (size_t)B_ * NL_ * 3;          // (B,NL,NP)

  // ---- workspace carve (256B aligned) ----
  char* wsp = (char*)d_ws;
  auto carve = [&](size_t bytes) {
    char* p = wsp;
    wsp += (bytes + 255) & ~(size_t)255;
    return p;
  };
  unsigned short* xLbf  = (unsigned short*)carve((size_t)B_ * NL_ * D_ * 2);
  unsigned short* xPbf  = (unsigned short*)carve((size_t)B_ * NP_ * D_ * 2);
  unsigned short* Wqbf  = (unsigned short*)carve((size_t)D_ * D_ * 2);
  unsigned short* Wkbf  = (unsigned short*)carve((size_t)D_ * D_ * 2);
  unsigned short* Wvbf  = (unsigned short*)carve((size_t)D_ * D_ * 2);
  unsigned short* s1p   = (unsigned short*)carve((size_t)D_ * DP_ * 2);
  unsigned short* s2bf  = (unsigned short*)carve((size_t)D_ * D_ * 2);
  unsigned short* Qbf   = (unsigned short*)carve((size_t)B_ * NL_ * D_ * 2);
  unsigned short* Kbf   = (unsigned short*)carve((size_t)B_ * NP_ * D_ * 2);
  unsigned short* Vt    = (unsigned short*)carve((size_t)B_ * D_ * NP_ * 2);
  float*          scor  = (float*)         carve((size_t)B_ * NL_ * NP_ * 4);
  unsigned short* prbf  = (unsigned short*)carve((size_t)B_ * NL_ * NP_ * 2);
  float*          Vagg  = (float*)         carve((size_t)B_ * NL_ * 3 * 4);
  unsigned short* scomb = (unsigned short*)carve((size_t)B_ * NL_ * DP_ * 2);
  unsigned short* Hbuf  = (unsigned short*)carve((size_t)B_ * NL_ * D_ * 2);

  // ---- 1. converts ----
  auto cvt = [&](const float* s, unsigned short* d, int n) {
    int blocks = (n + 255) / 256; if (blocks > 4096) blocks = 4096;
    k_f32_to_bf16<<<blocks, 256, 0, stream>>>(s, d, n);
  };
  cvt(xL, xLbf, B_ * NL_ * D_);
  cvt(xP, xPbf, B_ * NP_ * D_);
  cvt(Wq, Wqbf, D_ * D_);
  cvt(Wk, Wkbf, D_ * D_);
  cvt(Wv, Wvbf, D_ * D_);
  cvt(s2w, s2bf, D_ * D_);
  k_pad_s1<<<1088, 256, 0, stream>>>(s1w, s1p);

  // ---- 2. projections ----
  // grid = (N/64, M/256); block = 256 (8 waves, each a 32x64 tile)
  k_gemm_bf16out<<<dim3(D_ / 64, (B_ * NL_) / 256), 256, 0, stream>>>(
      xLbf, Wqbf, bq, Qbf, D_, D_, D_, D_);
  k_gemm_bf16out<<<dim3(D_ / 64, (B_ * NP_) / 256), 256, 0, stream>>>(
      xPbf, Wkbf, bk, Kbf, D_, D_, D_, D_);
  k_gemm_vT<<<dim3(D_ / 64, (B_ * NP_) / 256), 256, 0, stream>>>(
      xPbf, Wvbf, bv, Vt);

  // ---- 3. scores + mask ----
  k_gemm_scores<<<dim3(NP_ / 64, NL_ / 256, B_), 256, 0, stream>>>(
      Qbf, Kbf, posL, posP, scor);

  // ---- 4. softmax (probs f32 -> d_out, bf16 -> ws) ----
  k_softmax<<<B_ * NL_, 256, 0, stream>>>(scor, outP, prbf);

  // ---- 5. V_agg + v_norm column + pad ----
  k_vagg<<<B_ * NL_, 256, 0, stream>>>(outP, posL, posP, vprojw, Vagg, scomb);

  // ---- 6. s_context + residual -> s_comb cols 0..511 ----
  k_gemm_ctx<<<dim3(D_ / 64, NL_ / 256, B_), 256, 0, stream>>>(
      prbf, Vt, xL, scomb);

  // ---- 7. gate + V_out ----
  k_gate<<<B_ * NL_, 128, 0, stream>>>(scomb, gatew, gateb, vprojw, Vagg, outV);

  // ---- 8. GVP MLP ----
  k_gemm_silu<<<dim3(D_ / 64, (B_ * NL_) / 256), 256, 0, stream>>>(
      scomb, s1p, s1b, Hbuf);
  k_gemm_f32out<<<dim3(D_ / 64, (B_ * NL_) / 256), 256, 0, stream>>>(
      Hbuf, s2bf, s2b, outS);
}
